// Attention_25786983645954
// MI455X (gfx1250) — matrix operand order repaired and checked by reading, not compiled, not run on hardware
//
#include <hip/hip_runtime.h>
#include <hip/hip_bf16.h>

typedef __attribute__((ext_vector_type(16))) __bf16 v16bf;
typedef __attribute__((ext_vector_type(8)))  float  v8f;

typedef int v4i __attribute__((vector_size(16)));
typedef __attribute__((address_space(1))) v4i  as1_v4i;
typedef __attribute__((address_space(3))) v4i  as3_v4i;
typedef __attribute__((address_space(3))) char as3char;

typedef unsigned int u32x4 __attribute__((ext_vector_type(4)));
typedef int i32x4v __attribute__((ext_vector_type(4)));
typedef int i32x8v __attribute__((ext_vector_type(8)));

#define B_  2
#define S_  2048
#define D_  1024
#define H_  16
#define DH_ 64

#if __has_builtin(__builtin_amdgcn_global_load_async_to_lds_b128)
#define ASYNC_VIA_BUILTIN 1
#else
#define ASYNC_VIA_BUILTIN 0
#endif

#if __has_builtin(__builtin_amdgcn_tensor_load_to_lds)
#define HAVE_TDM 1
#else
#define HAVE_TDM 0
#endif

// 16-byte async copy global -> LDS (ASYNCcnt-tracked, no VGPR roundtrip)
__device__ __forceinline__ void async_cp16(__bf16* lds, const __bf16* g) {
#if ASYNC_VIA_BUILTIN
    __builtin_amdgcn_global_load_async_to_lds_b128(
        (as1_v4i*)g, (as3_v4i*)lds, 0, 0);
#else
    unsigned loff = (unsigned)(unsigned long long)(as3char*)lds;
    asm volatile("global_load_async_to_lds_b128 %0, %1, off"
                 :: "v"(loff), "v"(g) : "memory");
#endif
}

__device__ __forceinline__ void wait_async0() {
#if __has_builtin(__builtin_amdgcn_s_wait_asynccnt)
    __builtin_amdgcn_s_wait_asynccnt(0);
#else
    asm volatile("s_wait_asynccnt 0x0" ::: "memory");
#endif
}

__device__ __forceinline__ void wait_tensor0() {
#if __has_builtin(__builtin_amdgcn_s_wait_tensorcnt)
    __builtin_amdgcn_s_wait_tensorcnt(0);
#else
    asm volatile("s_wait_tensorcnt 0x0" ::: "memory");
#endif
}

// TDM 2D tile load: tile1 rows x tile0 elements (2B each), row stride stride0
// elements, into LDS at lds_addr with 16B pad after every 128B (rows of 72
// bf16 in LDS). D# per CDNA5 ISA §8.3/8.4; groups 2/3 zero (2D tensor).
__device__ __forceinline__ void tdm_load_2d(unsigned lds_addr, const void* gptr,
                                            unsigned tile0, unsigned tile1,
                                            unsigned long long stride0)
{
#if HAVE_TDM
    unsigned long long ga = (unsigned long long)gptr;
    u32x4 g0;
    g0.x = 1u;                                               // count=1 (valid)
    g0.y = lds_addr;                                         // LDS byte addr
    g0.z = (unsigned)ga;                                     // global[31:0]
    g0.w = (unsigned)((ga >> 32) & 0x01FFFFFFull) | (2u << 30); // global[56:32], type=2

    // group1: data_size=2B(1), pad_enable, pad_interval=32dw(4), pad_amount=4dw(3)
    unsigned flags = (1u << 16) | (1u << 20) | (4u << 22) | (3u << 25);
    unsigned long long td0 = tile0;                          // tensor_dim0
    unsigned long long td1 = 0x100000ull;                    // tensor_dim1: big
    unsigned long long qw0 = (unsigned long long)flags | ((td0 & 0xFFFFull) << 48);
    unsigned long long qw1 = ((td0 >> 16) & 0xFFFFull)
                           | ((td1 & 0xFFFFFFFFull) << 16)
                           | ((unsigned long long)(tile0 & 0xFFFFu) << 48);
    unsigned long long qw2 = (unsigned long long)(tile1 & 0xFFFFu)
                           | ((stride0 & 0xFFFFFFFFull) << 32);
    unsigned long long qw3 = (stride0 >> 32) & 0xFFFFull;

    i32x8v g1;
    g1[0] = (int)(unsigned)qw0;  g1[1] = (int)(unsigned)(qw0 >> 32);
    g1[2] = (int)(unsigned)qw1;  g1[3] = (int)(unsigned)(qw1 >> 32);
    g1[4] = (int)(unsigned)qw2;  g1[5] = (int)(unsigned)(qw2 >> 32);
    g1[6] = (int)(unsigned)qw3;  g1[7] = (int)(unsigned)(qw3 >> 32);

    i32x4v gz  = {0, 0, 0, 0};
    i32x8v gz8 = {0, 0, 0, 0, 0, 0, 0, 0};
    __builtin_amdgcn_tensor_load_to_lds(g0, g1, gz, gz, gz8, 0);
#else
    (void)lds_addr; (void)gptr; (void)tile0; (void)tile1; (void)stride0;
#endif
}

// ---------------------------------------------------------------------------
// fp32 -> bf16 conversion
// ---------------------------------------------------------------------------
__global__ void cvt_f32_bf16(const float* __restrict__ in, __bf16* __restrict__ out, int n) {
    int i = (blockIdx.x * blockDim.x + threadIdx.x) * 4;
    if (i + 3 < n) {
        float4 f = *(const float4*)(in + i);
        out[i + 0] = (__bf16)f.x;
        out[i + 1] = (__bf16)f.y;
        out[i + 2] = (__bf16)f.z;
        out[i + 3] = (__bf16)f.w;
    }
}

// W_O [H][D][DH] fp32 -> bf16 repacked [D][H*DH]
__global__ void cvt_wo(const float* __restrict__ in, __bf16* __restrict__ out) {
    int idx = blockIdx.x * blockDim.x + threadIdx.x;
    int dh = idx & (DH_ - 1);
    int dd = (idx >> 6) & (D_ - 1);
    int h  = idx >> 16;
    out[(size_t)dd * (H_ * DH_) + h * DH_ + dh] = (__bf16)in[idx];
}

// ---------------------------------------------------------------------------
// NT GEMM:  C[M,N] = A[M,K] * B[N,K]^T + bias[N]
//   mode 0: bf16 out, q/k layout [B,H,S,DH]
//   mode 1: f32 out, row-major [M,N]
//   mode 2: bf16 out, transposed v layout [B,H,DH,S]
// 128x128 tile, BK=64, double-buffered LDS, async global->LDS staging.
// ---------------------------------------------------------------------------
__global__ __launch_bounds__(256)
void gemm_nt(const __bf16* __restrict__ A,
             const __bf16* __restrict__ Bm,
             const float*  __restrict__ bias,
             __bf16* __restrict__ outb,
             float*  __restrict__ outf,
             int M, int N, int K, int mode)
{
    constexpr int LDT = 72;
    __shared__ __bf16 As[2][128][LDT];
    __shared__ __bf16 Bs[2][128][LDT];

    const int tid   = threadIdx.x;
    const int lane  = tid & 31;
    const int wid   = tid >> 5;
    const int wm    = wid >> 2;
    const int wn    = wid & 3;
    const int nl    = lane & 15;
    const int khalf = lane >> 4;

    const int m_blk = blockIdx.y * 128;
    const int n_blk = blockIdx.x * 128;

    const int row_ld = tid >> 1;
    const int qh     = tid & 1;

    v8f zero = {};
    v8f acc[4][2];
    for (int i = 0; i < 4; i++)
        for (int j = 0; j < 2; j++) acc[i][j] = zero;

    auto stage = [&](int buf, int k0) {
        const __bf16* ag = A  + (size_t)(m_blk + row_ld) * K + k0 + qh * 32;
        const __bf16* bg = Bm + (size_t)(n_blk + row_ld) * K + k0 + qh * 32;
#pragma unroll
        for (int u = 0; u < 4; u++) {
            async_cp16(&As[buf][row_ld][qh * 32 + u * 8], ag + u * 8);
            async_cp16(&Bs[buf][row_ld][qh * 32 + u * 8], bg + u * 8);
        }
    };

    auto compute = [&](int buf, int h32) {
        v16bf afr[4], bfr[2];
#pragma unroll
        for (int i = 0; i < 4; i++) {
            int m = wm * 64 + i * 16 + nl;
            uint4* d = (uint4*)&afr[i];
            d[0] = *(const uint4*)&As[buf][m][h32 + khalf * 16];
            d[1] = *(const uint4*)&As[buf][m][h32 + khalf * 16 + 8];
        }
#pragma unroll
        for (int j = 0; j < 2; j++) {
            int n = wn * 32 + j * 16 + nl;
            const uint4* p = (const uint4*)&Bs[buf][n][h32 + khalf * 16];
            uint4* d = (uint4*)&bfr[j];
            d[0] = p[0]; d[1] = p[1];
        }
#pragma unroll
        for (int i = 0; i < 4; i++)
#pragma unroll
            for (int j = 0; j < 2; j++)
                acc[i][j] = __builtin_amdgcn_wmma_f32_16x16x32_bf16(
                    false, afr[i], false, bfr[j], (short)0, acc[i][j], false, false);
    };

    const int nk = K >> 6;
    stage(0, 0);
    wait_async0();
    __syncthreads();
    for (int kt = 0; kt < nk; kt++) {
        const int cur = kt & 1;
        if (kt + 1 < nk) stage(cur ^ 1, (kt + 1) << 6);
        compute(cur, 0);
        compute(cur, 32);
        wait_async0();
        __syncthreads();
    }

    for (int i = 0; i < 4; i++) {
        int mg_base = m_blk + wm * 64 + i * 16 + khalf * 8;
        for (int j = 0; j < 2; j++) {
            int ng = n_blk + wn * 32 + j * 16 + nl;
            float bv = bias[ng];
            for (int r = 0; r < 8; r++) {
                int mg = mg_base + r;
                float v = acc[i][j][r] + bv;
                if (mode == 0) {
                    int b = mg >> 11, s = mg & (S_ - 1);
                    int h = ng >> 6,  dh = ng & (DH_ - 1);
                    outb[((size_t)(b * H_ + h) * S_ + s) * DH_ + dh] = (__bf16)v;
                } else if (mode == 2) {
                    int b = mg >> 11, s = mg & (S_ - 1);
                    int h = ng >> 6,  dh = ng & (DH_ - 1);
                    outb[((size_t)(b * H_ + h) * DH_ + dh) * S_ + s] = (__bf16)v;
                } else {
                    outf[(size_t)mg * N + ng] = v;
                }
            }
        }
    }
}

// ---------------------------------------------------------------------------
// Flash attention, causal. Block = (b, h, 64 q rows); 4 waves, 16 q rows each.
// K/V^T chunk staging via Tensor Data Mover (one wave issues 2 descriptors,
// TENSORcnt-tracked); double-buffered, 1 barrier per chunk.
// ---------------------------------------------------------------------------
__global__ __launch_bounds__(128)
void flash_attn(const __bf16* __restrict__ Q,
                const __bf16* __restrict__ Kb,
                const __bf16* __restrict__ VTg,
                __bf16* __restrict__ Z)
{
    constexpr int LDT = 72;
    __shared__ __bf16 Ks[2][64][LDT];           // [key][d]
    __shared__ __bf16 VTs[2][64][LDT];          // [d][key]
    __shared__ __bf16 Ps[4][16][LDT];           // per-wave P tile [qrow][key]

    const int tid   = threadIdx.x;
    const int lane  = tid & 31;
    const int w     = tid >> 5;
    const int nl    = lane & 15;
    const int khalf = lane >> 4;

    const int qt = blockIdx.x;
    const int h  = blockIdx.y;
    const int b  = blockIdx.z;
    const int q0 = qt * 64;

    const __bf16* qp  = Q   + ((size_t)b * H_ + h) * S_ * DH_;
    const __bf16* kp  = Kb  + ((size_t)b * H_ + h) * S_ * DH_;
    const __bf16* vth = VTg + ((size_t)b * H_ + h) * DH_ * S_;

    v16bf qfr[2];
    {
        const __bf16* qrow = qp + (size_t)(q0 + w * 16 + nl) * DH_;
        for (int t = 0; t < 2; t++) {
            uint4* d = (uint4*)&qfr[t];
            d[0] = *(const uint4*)(qrow + t * 32 + khalf * 16);
            d[1] = *(const uint4*)(qrow + t * 32 + khalf * 16 + 8);
        }
    }

    float mrow[8], lrow[8];
    v8f zerov = {};
    v8f o_acc[4];
    for (int r = 0; r < 8; r++) { mrow[r] = -3.0e38f; lrow[r] = 0.0f; }
    for (int dt = 0; dt < 4; dt++) o_acc[dt] = zerov;

    const int srow = tid >> 1;
    const int qh   = tid & 1;

    auto stage = [&](int buf, int k0) {
#if HAVE_TDM
        if (w == 0) {   // one wave issues both tile descriptors (EXEC ignored)
            unsigned lds_k = (unsigned)(unsigned long long)(as3char*)&Ks[buf][0][0];
            unsigned lds_v = (unsigned)(unsigned long long)(as3char*)&VTs[buf][0][0];
            tdm_load_2d(lds_k, kp + (size_t)k0 * DH_, 64, 64, DH_);  // K rows
            tdm_load_2d(lds_v, vth + k0,              64, 64, S_);   // V^T rows
        }
#else
        const __bf16* kg = kp  + (size_t)(k0 + srow) * DH_ + qh * 32;
        const __bf16* vg = vth + (size_t)srow * S_ + k0 + qh * 32;
#pragma unroll
        for (int u = 0; u < 4; u++) {
            async_cp16(&Ks[buf][srow][qh * 32 + u * 8],  kg + u * 8);
            async_cp16(&VTs[buf][srow][qh * 32 + u * 8], vg + u * 8);
        }
#endif
    };

    stage(0, 0);
    wait_tensor0();
    wait_async0();
    __syncthreads();

    for (int jc = 0; jc <= qt; ++jc) {
        const int k0  = jc * 64;
        const int cur = jc & 1;
        if (jc < qt) stage(cur ^ 1, k0 + 64);            // overlap DMA with math
        if (jc + 2 <= qt) {                              // L2 prefetch 2 ahead
            __builtin_prefetch(kp  + (size_t)(k0 + 128 + srow) * DH_, 0, 1);
            __builtin_prefetch(vth + (size_t)srow * S_ + k0 + 128, 0, 1);
        }

        // ---- scores: 16(q) x 64(k) per wave ----
        v8f sc[4];
        for (int kn = 0; kn < 4; kn++) {
            v8f a = zerov;
            for (int t = 0; t < 2; t++) {
                v16bf bf;
                const uint4* p = (const uint4*)&Ks[cur][kn * 16 + nl][t * 32 + khalf * 16];
                uint4* d = (uint4*)&bf;
                d[0] = p[0]; d[1] = p[1];
                a = __builtin_amdgcn_wmma_f32_16x16x32_bf16(
                        false, qfr[t], false, bf, (short)0, a, false, false);
            }
            sc[kn] = a;
        }

        // ---- scale, causal mask, online softmax ----
        const bool diag = (jc == qt);
        float p[4][8];
        for (int r = 0; r < 8; r++) {
            int mg = q0 + w * 16 + khalf * 8 + r;
            float mx = -3.0e38f;
            for (int kn = 0; kn < 4; kn++) {
                float s = sc[kn][r] * 0.125f;
                if (diag && (k0 + kn * 16 + nl) > mg) s = -3.0e38f;
                p[kn][r] = s;
                mx = fmaxf(mx, s);
            }
            for (int off = 1; off < 16; off <<= 1)
                mx = fmaxf(mx, __shfl_xor(mx, off, 32));
            float mnew  = fmaxf(mrow[r], mx);
            float alpha = __expf(mrow[r] - mnew);
            float rs = 0.0f;
            for (int kn = 0; kn < 4; kn++) {
                float e = __expf(p[kn][r] - mnew);
                p[kn][r] = e;
                rs += e;
            }
            for (int off = 1; off < 16; off <<= 1)
                rs += __shfl_xor(rs, off, 32);
            lrow[r] = lrow[r] * alpha + rs;
            mrow[r] = mnew;
            for (int dt = 0; dt < 4; dt++) o_acc[dt][r] *= alpha;
        }

        // ---- P (C layout) -> bf16 A layout, wave-local LDS tile ----
        for (int kn = 0; kn < 4; kn++)
            for (int r = 0; r < 8; r++)
                Ps[w][khalf * 8 + r][kn * 16 + nl] = (__bf16)p[kn][r];

        // ---- O += P @ V ----
        for (int t = 0; t < 2; t++) {
            v16bf af;
            const __bf16* prow = &Ps[w][nl][0];
            uint4* da = (uint4*)&af;
            da[0] = *(const uint4*)(prow + t * 32 + khalf * 16);
            da[1] = *(const uint4*)(prow + t * 32 + khalf * 16 + 8);
            for (int dt = 0; dt < 4; dt++) {
                v16bf bf;
                const uint4* pb = (const uint4*)&VTs[cur][dt * 16 + nl][t * 32 + khalf * 16];
                uint4* db = (uint4*)&bf;
                db[0] = pb[0]; db[1] = pb[1];
                o_acc[dt] = __builtin_amdgcn_wmma_f32_16x16x32_bf16(
                        false, af, false, bf, (short)0, o_acc[dt], false, false);
            }
        }

        wait_tensor0();      // next chunk's TDM transfers landed in LDS
        wait_async0();       // (fallback path)
        __syncthreads();     // all waves done reading cur before it is reused
    }

    // ---- normalize and store z: [B,S,H*DH] ----
    for (int r = 0; r < 8; r++) {
        float inv = 1.0f / lrow[r];
        int sg = q0 + w * 16 + khalf * 8 + r;
        __bf16* zr = Z + ((size_t)b * S_ + sg) * (H_ * DH_) + h * DH_;
        for (int dt = 0; dt < 4; dt++)
            zr[dt * 16 + nl] = (__bf16)(o_acc[dt][r] * inv);
    }
}

// ---------------------------------------------------------------------------
extern "C" void kernel_launch(void* const* d_in, const int* in_sizes, int n_in,
                              void* d_out, int out_size, void* d_ws, size_t ws_size,
                              hipStream_t stream) {
    (void)in_sizes; (void)n_in; (void)out_size; (void)ws_size;
    const float* x   = (const float*)d_in[0];
    const float* W_Q = (const float*)d_in[1];
    const float* W_K = (const float*)d_in[2];
    const float* W_V = (const float*)d_in[3];
    const float* W_O = (const float*)d_in[4];
    const float* b_Q = (const float*)d_in[5];
    const float* b_K = (const float*)d_in[6];
    const float* b_V = (const float*)d_in[7];
    const float* b_O = (const float*)d_in[8];
    float* out = (float*)d_out;

    char* ws = (char*)d_ws;
    const size_t nx   = (size_t)B_ * S_ * D_;
    const size_t nw   = (size_t)H_ * DH_ * D_;
    const size_t nqkv = (size_t)B_ * H_ * S_ * DH_;
    __bf16* xb  = (__bf16*)ws; ws += nx   * 2;
    __bf16* wq  = (__bf16*)ws; ws += nw   * 2;
    __bf16* wk  = (__bf16*)ws; ws += nw   * 2;
    __bf16* wv  = (__bf16*)ws; ws += nw   * 2;
    __bf16* wo  = (__bf16*)ws; ws += nw   * 2;
    __bf16* qb  = (__bf16*)ws; ws += nqkv * 2;
    __bf16* kb  = (__bf16*)ws; ws += nqkv * 2;
    __bf16* vtb = (__bf16*)ws; ws += nqkv * 2;   // [B,H,DH,S]
    __bf16* zb  = (__bf16*)ws; ws += nqkv * 2;

    cvt_f32_bf16<<<(int)(nx / 4 / 256), 256, 0, stream>>>(x, xb, (int)nx);
    cvt_f32_bf16<<<(int)(nw / 4 / 256), 256, 0, stream>>>(W_Q, wq, (int)nw);
    cvt_f32_bf16<<<(int)(nw / 4 / 256), 256, 0, stream>>>(W_K, wk, (int)nw);
    cvt_f32_bf16<<<(int)(nw / 4 / 256), 256, 0, stream>>>(W_V, wv, (int)nw);
    cvt_wo<<<(int)(nw / 256), 256, 0, stream>>>(W_O, wo);

    dim3 gg(D_ / 128, (B_ * S_) / 128);
    gemm_nt<<<gg, 256, 0, stream>>>(xb, wq, b_Q, qb,  nullptr, B_ * S_, H_ * DH_, D_, 0);
    gemm_nt<<<gg, 256, 0, stream>>>(xb, wk, b_K, kb,  nullptr, B_ * S_, H_ * DH_, D_, 0);
    gemm_nt<<<gg, 256, 0, stream>>>(xb, wv, b_V, vtb, nullptr, B_ * S_, H_ * DH_, D_, 2);

    dim3 fg(S_ / 64, H_, B_);
    flash_attn<<<fg, 128, 0, stream>>>(qb, kb, vtb, zb);

    gemm_nt<<<gg, 256, 0, stream>>>(zb, wo, b_O, nullptr, out, B_ * S_, D_, H_ * DH_, 1);
}
